// KNRMModel_37194416783577
// MI455X (gfx1250) — compile-verified, matrix operand order checked by reading, not run on hardware
//
#include <hip/hip_runtime.h>
#include <hip/hip_bf16.h>

typedef __attribute__((ext_vector_type(16))) _Float16 v16h;
typedef __attribute__((ext_vector_type(8)))  float    v8f;

#define B_   32
#define C_   5
#define H_   50
#define S_   20
#define E_   300
#define K_   20
#define EPAD 320   // K padded for 10x wmma 16x16x32 steps
#define SPAD 32    // M/N padded to two 16-tiles

// ---------------------------------------------------------------- zero ----
__global__ void knrm_zero_kernel(float* acc, int n) {
    int i = blockIdx.x * blockDim.x + threadIdx.x;
    if (i < n) acc[i] = 0.0f;
}

// ----------------------------------------------- gather + l2norm -> f16 ----
// One wave (32 lanes) per padded tile row. Rows [0, B*C*SPAD) -> cddN,
// rest -> hisN. Pad rows (s>=20) and pad cols (e>=300) are zero-filled so
// the WMMA kernel can load fragments directly with no bounds checks.
__global__ __launch_bounds__(32) void knrm_normalize_kernel(
    const int* __restrict__ cand, const int* __restrict__ clicked,
    const float* __restrict__ emb,
    _Float16* __restrict__ cddN, _Float16* __restrict__ hisN) {
    int row  = blockIdx.x;
    int lane = threadIdx.x;
    const int cddRows = B_ * C_ * SPAD;

    _Float16* dst;
    int token = -1;
    if (row < cddRows) {
        int tile = row / SPAD, s = row % SPAD;
        dst = cddN + (size_t)row * EPAD;
        if (s < S_) token = cand[tile * S_ + s];
    } else {
        int r2 = row - cddRows;
        int tile = r2 / SPAD, s = r2 % SPAD;
        dst = hisN + (size_t)r2 * EPAD;
        if (s < S_) token = clicked[tile * S_ + s];
    }

    if (token < 0) {  // padding row
        for (int e = lane; e < EPAD; e += 32) dst[e] = (_Float16)0.0f;
        return;
    }

    const float* src = emb + (size_t)token * E_;
    float v[10];
    float ss = 0.0f;
    int n = 0;
    for (int e = lane; e < E_; e += 32) { float x = src[e]; v[n++] = x; ss += x * x; }
    // wave32 butterfly reduction
    for (int off = 16; off > 0; off >>= 1) ss += __shfl_xor(ss, off, 32);
    float r = 1.0f / fmaxf(sqrtf(ss), 1e-12f);
    n = 0;
    for (int e = lane; e < EPAD; e += 32) {
        float x = (e < E_) ? v[n++] * r : 0.0f;
        dst[e] = (_Float16)x;
    }
}

// -------------------------------- WMMA sim + RBF pooling + score accum ----
// One 32-lane block per (b,c,h). sim (20x20 padded to 32x32) via 2x2 tiles
// of v_wmma_f32_16x16x32_f16 over K=320 (10 steps, 40 wmma per block).
// Fragment fetch: lane L loads 16 consecutive K-halves of row (L%16) at
// K-offset (L/16)*16 -- one aligned 32B global load per fragment, for both
// the A (16x32) and B (32x16) VGPR layouts.
__global__ __launch_bounds__(32) void knrm_simpool_kernel(
    const _Float16* __restrict__ cddN, const _Float16* __restrict__ hisN,
    const float* __restrict__ padC, const float* __restrict__ padH,
    const float* __restrict__ ltr_w, float* __restrict__ scoreAcc) {
    int bch = blockIdx.x;
    int b = bch / (C_ * H_);
    int rem = bch % (C_ * H_);
    int c = rem / H_;
    int h = rem % H_;
    int lane = threadIdx.x;

    const _Float16* A  = cddN + (size_t)(b * C_ + c) * SPAD * EPAD;
    const _Float16* Bm = hisN + (size_t)(b * H_ + h) * SPAD * EPAD;
    int lr = lane & 15;           // row within 16-tile
    int lk = (lane >> 4) << 4;    // K sub-offset (0 or 16)

    v8f acc00 = {}, acc01 = {}, acc10 = {}, acc11 = {};
    for (int kk = 0; kk < EPAD; kk += 32) {
        int ko = kk + lk;
        v16h a0 = *(const v16h*)(A  + (size_t)lr * EPAD + ko);
        v16h a1 = *(const v16h*)(A  + (size_t)(lr + 16) * EPAD + ko);
        v16h b0 = *(const v16h*)(Bm + (size_t)lr * EPAD + ko);
        v16h b1 = *(const v16h*)(Bm + (size_t)(lr + 16) * EPAD + ko);
        acc00 = __builtin_amdgcn_wmma_f32_16x16x32_f16(false, a0, false, b0, (short)0, acc00, false, false);
        acc01 = __builtin_amdgcn_wmma_f32_16x16x32_f16(false, a0, false, b1, (short)0, acc01, false, false);
        acc10 = __builtin_amdgcn_wmma_f32_16x16x32_f16(false, a1, false, b0, (short)0, acc10, false, false);
        acc11 = __builtin_amdgcn_wmma_f32_16x16x32_f16(false, a1, false, b1, (short)0, acc11, false, false);
    }

    // Spill sim to LDS. C/D layout: VGPR v -> M = v + 8*(lane>=16), N = lane%16.
    __shared__ float simLds[SPAD][SPAD + 1];
    __shared__ float padHl[S_];
    int mofs = (lane >> 4) * 8;
    int nofs = lane & 15;
#pragma unroll
    for (int v = 0; v < 8; ++v) {
        simLds[mofs + v][nofs]            = acc00[v];
        simLds[mofs + v][16 + nofs]       = acc01[v];
        simLds[16 + mofs + v][nofs]       = acc10[v];
        simLds[16 + mofs + v][16 + nofs]  = acc11[v];
    }
    if (lane < S_) padHl[lane] = padH[(b * H_ + h) * S_ + lane];
    __syncthreads();

    // RBF kernel pooling: 400 (s,k) pairs over 32 lanes, 20 exps each.
    float contrib = 0.0f;
    for (int p = lane; p < S_ * K_; p += 32) {
        int s = p / K_;
        int k = p - s * K_;
        float mu  = -0.9f + 0.1f * (float)k;
        float i2s = (k == K_ - 1) ? 5.0e5f : 50.0f;  // 1/(2*sigma^2)
        float sum = 0.0f;
#pragma unroll 4
        for (int t = 0; t < S_; ++t) {
            float d = simLds[s][t] - mu;
            sum += __expf(-d * d * i2s) * padHl[t];
        }
        float lp = __logf(fmaxf(sum, 1e-10f)) * 0.01f * padC[(b * C_ + c) * S_ + s];
        contrib += lp * ltr_w[h * K_ + k];
    }
    for (int off = 16; off > 0; off >>= 1) contrib += __shfl_xor(contrib, off, 32);
    if (lane == 0) atomicAdd(&scoreAcc[b * C_ + c], contrib);
}

// ------------------------------------------- bias + log_softmax over C ----
__global__ void knrm_finalize_kernel(const float* __restrict__ scoreAcc,
                                     const float* __restrict__ ltr_b,
                                     float* __restrict__ out) {
    int b = threadIdx.x;
    if (b >= B_) return;
    float x[C_];
    float m = -1e30f;
    for (int c = 0; c < C_; ++c) { x[c] = scoreAcc[b * C_ + c] + ltr_b[0]; m = fmaxf(m, x[c]); }
    float sum = 0.0f;
    for (int c = 0; c < C_; ++c) sum += expf(x[c] - m);
    float l = logf(sum);
    for (int c = 0; c < C_; ++c) out[b * C_ + c] = x[c] - m - l;
}

extern "C" void kernel_launch(void* const* d_in, const int* in_sizes, int n_in,
                              void* d_out, int out_size, void* d_ws, size_t ws_size,
                              hipStream_t stream) {
    (void)in_sizes; (void)n_in; (void)out_size; (void)ws_size;
    const int*   cand    = (const int*)d_in[0];    // (B,C,S) int32
    const int*   clicked = (const int*)d_in[1];    // (B,H,S) int32
    const float* padC    = (const float*)d_in[2];  // (B,C,S) f32
    const float* padH    = (const float*)d_in[3];  // (B,H,S) f32
    const float* emb     = (const float*)d_in[4];  // (V,E)   f32
    const float* ltr_w   = (const float*)d_in[5];  // (H*K,1) f32
    const float* ltr_b   = (const float*)d_in[6];  // (1,)    f32
    float* out = (float*)d_out;

    // Workspace layout (all offsets 32B-aligned for v16h loads):
    //   [0, 640)                    : score accumulator (B*C floats)
    //   [1024, +3,276,800)          : cddN  (B*C tiles of 32x320 f16)
    //   [.., +32,768,000)           : hisN  (B*H tiles of 32x320 f16)
    char* ws = (char*)d_ws;
    float*    scoreAcc = (float*)ws;
    _Float16* cddN = (_Float16*)(ws + 1024);
    _Float16* hisN = (_Float16*)(ws + 1024 + (size_t)B_ * C_ * SPAD * EPAD * 2);

    knrm_zero_kernel<<<1, 256, 0, stream>>>(scoreAcc, B_ * C_);

    int rows = (B_ * C_ + B_ * H_) * SPAD;  // 56,320 tile rows
    knrm_normalize_kernel<<<rows, 32, 0, stream>>>(cand, clicked, emb, cddN, hisN);

    knrm_simpool_kernel<<<B_ * C_ * H_, 32, 0, stream>>>(cddN, hisN, padC, padH,
                                                         ltr_w, scoreAcc);

    knrm_finalize_kernel<<<1, 32, 0, stream>>>(scoreAcc, ltr_b, out);
}
